// StereoLayers_39273180955381
// MI455X (gfx1250) — compile-verified, matrix operand order checked by reading, not run on hardware
//
#include <hip/hip_runtime.h>
#include <hip/hip_bf16.h>
#include <math.h>

// ---------------------------------------------------------------------------
// Problem constants (B=4, L=8192, C=256, HEADS=1, FFN_HID=2048)
// ---------------------------------------------------------------------------
#define LTOT   8192
#define CDIM   256
#define BBATCH 4
#define MTOT   (BBATCH * LTOT)   // 32768
#define FFNH   2048
#define EPSL   1e-5f

// CDNA5 async global->LDS staging (inline asm; flip to 0 for sync fallback)
#define USE_ASYNC 1

typedef __attribute__((ext_vector_type(16))) __bf16 v16bf;
typedef __attribute__((ext_vector_type(8)))  float  v8f;

union ABf {
    v16bf          v;
    unsigned int   u32[8];
    unsigned short us[16];
};
union U16x8 { uint4 v; unsigned short s[8]; };

__device__ __forceinline__ unsigned short f2bf(float f) {
    union { float f; unsigned u; } x; x.f = f;
    unsigned u = x.u;
    unsigned r = u + 0x7FFFu + ((u >> 16) & 1u);   // round-to-nearest-even
    return (unsigned short)(r >> 16);
}

// 16B copy global -> LDS (async DMA on CDNA5, tracked by ASYNCcnt)
__device__ __forceinline__ void copy16(unsigned short* lds,
                                       const unsigned short* g) {
#if USE_ASYNC
    unsigned off = (unsigned)(unsigned long long)lds;   // LDS byte address
    asm volatile("global_load_async_to_lds_b128 %0, %1, off"
                 :: "v"(off), "v"(g) : "memory");
#else
    *(uint4*)lds = *(const uint4*)g;
#endif
}

__device__ __forceinline__ void wait_async() {
#if USE_ASYNC
    asm volatile("s_wait_asynccnt 0x0" ::: "memory");
#endif
}

// Padded LDS row stride (elements): 40 -> 80B rows, bank-spread, 16B aligned
#define LDS_STRIDE 40

// ---------------------------------------------------------------------------
// bf16 WMMA GEMM, double-buffered async LDS staging.
//   C[m,n] = sum_k A[m,k] * B[n,k]     (B pre-transposed: [N][K] row-major)
// Block = 256 threads = 8 waves (4 M x 2 N). Block tile 128(M) x 64(N), K=32.
// Wave tile 32x32: 4 WMMAs / step, 2 A-frags x 2 B-frags (quadratic reuse).
// grid = (N/64, M/128, batches*splitK)
// STORE: 0 = f32, 1 = bf16 row-major, 2 = bf16 transposed per-batch (q^T)
// ---------------------------------------------------------------------------
template <int CONCAT, int EPI, int SPLIT, int STORE>
__global__ __launch_bounds__(256)
void gemm_bf16(const unsigned short* __restrict__ A,
               const unsigned short* __restrict__ A2,
               const unsigned short* __restrict__ B,
               const float* __restrict__ bias,
               void* __restrict__ Cptr,
               int M, int N, int K, int lda, int lda2, int ldb, int ldc,
               long long strideA, long long strideB, long long strideC,
               int splitK, long long splitStride, int Asplit)
{
    __shared__ __attribute__((aligned(16))) unsigned short Asm[2][128 * LDS_STRIDE];
    __shared__ __attribute__((aligned(16))) unsigned short Bsm[2][64 * LDS_STRIDE];

    const int tid  = threadIdx.x;
    const int lane = tid & 31;
    const int wave = tid >> 5;
    const int half = lane >> 4;
    const int r15  = lane & 15;

    const int bz    = blockIdx.z;
    const int batch = bz / splitK;
    const int split = bz % splitK;
    const int kPer  = K / splitK;
    const int kBeg  = split * kPer;

    const int mBase = blockIdx.y * 128;
    const int nBase = blockIdx.x * 64;

    const long long aBat = (long long)batch * strideA;
    const long long bBat = (long long)batch * strideB;

    // staging: A tile 128x32 (2 segs/thread), B tile 64x32 (1 seg/thread)
    const int aRow = tid >> 1;            // 0..127
    const int aSeg = (tid & 1) * 16;      // 0 or 16 (plus +8 for 2nd seg)
    const int bRow = tid >> 2;            // 0..63
    const int bSeg = (tid & 3) * 8;       // 0,8,16,24

    const int mFrag = (wave >> 1) * 32;   // 0,32,64,96
    const int nFrag = (wave & 1) * 32;    // 0,32

    v8f acc[2][2] = {{{}, {}}, {{}, {}}};

    auto stage = [&](int k0, int buf) {
        // A segments (two 16B segs, same row)
        const int gk = k0 + aSeg;
        const unsigned short *pa0, *pa1;
        if (CONCAT) {
            pa0 = (gk < Asplit)
                ? (A  + (long long)(mBase + aRow) * lda  + gk)
                : (A2 + (long long)(mBase + aRow) * lda2 + (gk - Asplit));
            pa1 = (gk + 8 < Asplit)
                ? (A  + (long long)(mBase + aRow) * lda  + gk + 8)
                : (A2 + (long long)(mBase + aRow) * lda2 + (gk + 8 - Asplit));
        } else {
            pa0 = A + aBat + (long long)(mBase + aRow) * lda + gk;
            pa1 = pa0 + 8;
        }
        copy16(&Asm[buf][aRow * LDS_STRIDE + aSeg],     pa0);
        copy16(&Asm[buf][aRow * LDS_STRIDE + aSeg + 8], pa1);
        // B segment
        const unsigned short* pb =
            B + bBat + (long long)(nBase + bRow) * ldb + (k0 + bSeg);
        copy16(&Bsm[buf][bRow * LDS_STRIDE + bSeg], pb);
    };

    const int nSteps = kPer >> 5;
    stage(kBeg, 0);                        // prologue

    for (int s = 0; s < nSteps; ++s) {
        const int buf = s & 1;
        wait_async();          // own async writes done (incl. next-tile issue)
        __syncthreads();       // everyone's writes for tile s are visible
        if (s + 1 < nSteps) stage(kBeg + (s + 1) * 32, buf ^ 1);

        ABf a0, a1, b0, b1;
#pragma unroll
        for (int i = 0; i < 8; ++i) {
            const int kk = 2 * i + ((i >= 4) ? 8 : 0) + half * 8;  // even
            a0.u32[i] = *(const unsigned int*)&Asm[buf][(mFrag + r15) * LDS_STRIDE + kk];
            a1.u32[i] = *(const unsigned int*)&Asm[buf][(mFrag + 16 + r15) * LDS_STRIDE + kk];
            b0.u32[i] = *(const unsigned int*)&Bsm[buf][(nFrag + r15) * LDS_STRIDE + kk];
            b1.u32[i] = *(const unsigned int*)&Bsm[buf][(nFrag + 16 + r15) * LDS_STRIDE + kk];
        }
        acc[0][0] = __builtin_amdgcn_wmma_f32_16x16x32_bf16(
            false, a0.v, false, b0.v, (short)0, acc[0][0], false, false);
        acc[0][1] = __builtin_amdgcn_wmma_f32_16x16x32_bf16(
            false, a0.v, false, b1.v, (short)0, acc[0][1], false, false);
        acc[1][0] = __builtin_amdgcn_wmma_f32_16x16x32_bf16(
            false, a1.v, false, b0.v, (short)0, acc[1][0], false, false);
        acc[1][1] = __builtin_amdgcn_wmma_f32_16x16x32_bf16(
            false, a1.v, false, b1.v, (short)0, acc[1][1], false, false);
    }

    // ---- epilogue / store ----
    float*          Cf = (float*)Cptr;
    unsigned short* Cb = (unsigned short*)Cptr;
#pragma unroll
    for (int t = 0; t < 2; ++t) {
#pragma unroll
        for (int u = 0; u < 2; ++u) {
            const int n = nBase + nFrag + u * 16 + r15;
#pragma unroll
            for (int rr = 0; rr < 8; ++rr) {
                const int m = mBase + mFrag + t * 16 + rr + half * 8;
                float vv = acc[t][u][rr];
                if (EPI >= 1) vv += bias[n];
                if (EPI == 2)
                    vv = 0.5f * vv * (1.0f + erff(vv * 0.70710678118654752f));
                if (SPLIT) {
                    Cf[(long long)split * splitStride + (long long)batch * strideC
                       + (long long)m * ldc + n] = vv;
                } else if (STORE == 1) {
                    Cb[(long long)batch * strideC + (long long)m * ldc + n] = f2bf(vv);
                } else if (STORE == 2) {
                    // per-batch transpose: out[b][n][tok], m = b*LTOT + tok
                    const long long bb  = (long long)(m >> 13);
                    const long long tok = (long long)(m & (LTOT - 1));
                    Cb[(bb * CDIM + n) * LTOT + tok] = f2bf(vv);
                } else {
                    Cf[(long long)batch * strideC + (long long)m * ldc + n] = vv;
                }
            }
        }
    }
}

// ---------------------------------------------------------------------------
// Split-K reduction -> bf16 [N][K] context matrix
// ---------------------------------------------------------------------------
__global__ void reduce_split_bf16(const float* __restrict__ part,
                                  unsigned short* __restrict__ out,
                                  long long n, int splits)
{
    long long i = (long long)blockIdx.x * blockDim.x + threadIdx.x;
    if (i < n) {
        float s = 0.f;
        for (int j = 0; j < splits; ++j) s += part[(long long)j * n + i];
        out[i] = f2bf(s);
    }
}

// ---------------------------------------------------------------------------
// fp32 -> bf16 flat conversion (8 elements / thread, b128 in / b128 out)
// ---------------------------------------------------------------------------
__global__ __launch_bounds__(256)
void cvt_bf16(const float* __restrict__ in, unsigned short* __restrict__ out,
              long long n)
{
    long long i = ((long long)blockIdx.x * blockDim.x + threadIdx.x) * 8;
    if (i + 8 <= n) {
        float4 f0 = *(const float4*)(in + i);
        float4 f1 = *(const float4*)(in + i + 4);
        U16x8 o;
        o.s[0] = f2bf(f0.x); o.s[1] = f2bf(f0.y);
        o.s[2] = f2bf(f0.z); o.s[3] = f2bf(f0.w);
        o.s[4] = f2bf(f1.x); o.s[5] = f2bf(f1.y);
        o.s[6] = f2bf(f1.z); o.s[7] = f2bf(f1.w);
        *(uint4*)(out + i) = o.v;
    }
}

// ---------------------------------------------------------------------------
// fp32 [K][N] -> bf16 [N][K] transpose-convert (weights), 32x32 LDS tiles
// ---------------------------------------------------------------------------
__global__ __launch_bounds__(256)
void cvt_bf16_t(const float* __restrict__ in, unsigned short* __restrict__ out,
                int K, int N)
{
    __shared__ float tile[32][33];
    const int k0 = blockIdx.x * 32, n0 = blockIdx.y * 32;
    const int tx = threadIdx.x & 31, ty = threadIdx.x >> 5;  // 32 x 8
#pragma unroll
    for (int r = 0; r < 32; r += 8)
        tile[ty + r][tx] = in[(long long)(k0 + ty + r) * N + (n0 + tx)];
    __syncthreads();
#pragma unroll
    for (int r = 0; r < 32; r += 8)
        out[(long long)(n0 + ty + r) * K + (k0 + tx)] = f2bf(tile[tx][ty + r]);
}

// ---------------------------------------------------------------------------
// Column softmax over token dim (axis=1), 3-pass online. pass3 emits bf16 k^T.
// ---------------------------------------------------------------------------
#define SM_CHUNKS 32
#define SM_ROWS   (LTOT / SM_CHUNKS)   // 256

__global__ __launch_bounds__(256)
void softmax_pass1(const float* __restrict__ k, float* __restrict__ pm,
                   float* __restrict__ ps)
{
    const int b = blockIdx.x, ch = blockIdx.y, c = threadIdx.x;
    const float* base = k + (long long)b * LTOT * CDIM
                          + (long long)ch * SM_ROWS * CDIM + c;
    float m = -INFINITY, s = 0.f;
    for (int r = 0; r < SM_ROWS; ++r) {
        float x  = base[(long long)r * CDIM];
        float nm = fmaxf(m, x);
        s = s * __expf(m - nm) + __expf(x - nm);
        m = nm;
    }
    pm[((long long)b * SM_CHUNKS + ch) * CDIM + c] = m;
    ps[((long long)b * SM_CHUNKS + ch) * CDIM + c] = s;
}

__global__ __launch_bounds__(256)
void softmax_pass2(const float* __restrict__ pm, const float* __restrict__ ps,
                   float* __restrict__ gm, float* __restrict__ gi)
{
    const int b = blockIdx.x, c = threadIdx.x;
    float m = -INFINITY;
    for (int ch = 0; ch < SM_CHUNKS; ++ch)
        m = fmaxf(m, pm[((long long)b * SM_CHUNKS + ch) * CDIM + c]);
    float s = 0.f;
    for (int ch = 0; ch < SM_CHUNKS; ++ch)
        s += ps[((long long)b * SM_CHUNKS + ch) * CDIM + c] *
             __expf(pm[((long long)b * SM_CHUNKS + ch) * CDIM + c] - m);
    gm[b * CDIM + c] = m;
    gi[b * CDIM + c] = 1.0f / s;
}

// writes kt[b][c][token] in bf16 ([N][K] layout for the context GEMM)
__global__ __launch_bounds__(256)
void softmax_pass3(const float* __restrict__ k, const float* __restrict__ gm,
                   const float* __restrict__ gi, unsigned short* __restrict__ kt)
{
    const int b = blockIdx.x, ch = blockIdx.y, c = threadIdx.x;
    const float m   = gm[b * CDIM + c];
    const float inv = gi[b * CDIM + c];
    const float* base = k + (long long)b * LTOT * CDIM
                          + (long long)ch * SM_ROWS * CDIM + c;
    unsigned short* ob = kt + ((long long)b * CDIM + c) * LTOT
                            + (long long)ch * SM_ROWS;
    for (int r = 0; r < SM_ROWS; ++r)
        ob[r] = f2bf(__expf(base[(long long)r * CDIM] - m) * inv);
}

// ---------------------------------------------------------------------------
// Row LayerNorm (C=256), optional fused residual add, optional bf16 output.
// ---------------------------------------------------------------------------
template <int OUT_BF16, int ADD_SRC>
__global__ __launch_bounds__(256)
void layernorm_k(const float* __restrict__ x, const float* __restrict__ gamma,
                 const float* __restrict__ beta, const float* __restrict__ src,
                 void* __restrict__ yv)
{
    __shared__ float sred[256];
    const long long row = blockIdx.x;
    const int c = threadIdx.x;
    const float v = x[row * CDIM + c];

    sred[c] = v; __syncthreads();
    for (int o = 128; o > 0; o >>= 1) {
        if (c < o) sred[c] += sred[c + o];
        __syncthreads();
    }
    const float mu = sred[0] * (1.0f / CDIM);
    __syncthreads();

    const float d = v - mu;
    sred[c] = d * d; __syncthreads();
    for (int o = 128; o > 0; o >>= 1) {
        if (c < o) sred[c] += sred[c + o];
        __syncthreads();
    }
    const float var = sred[0] * (1.0f / CDIM);

    float out = d * rsqrtf(var + EPSL) * gamma[c] + beta[c];
    if (ADD_SRC) out += src[row * CDIM + c];
    if (OUT_BF16) ((unsigned short*)yv)[row * CDIM + c] = f2bf(out);
    else          ((float*)yv)[row * CDIM + c] = out;
}

// ---------------------------------------------------------------------------
// Host orchestration
// ---------------------------------------------------------------------------
extern "C" void kernel_launch(void* const* d_in, const int* in_sizes, int n_in,
                              void* d_out, int out_size, void* d_ws, size_t ws_size,
                              hipStream_t stream)
{
    (void)in_sizes; (void)n_in; (void)out_size; (void)ws_size;

    const float* src = (const float*)d_in[0];
    const float* tgt = (const float*)d_in[1];
    const float* Wq  = (const float*)d_in[2];
    const float* Wk  = (const float*)d_in[3];
    const float* Wv  = (const float*)d_in[4];
    const float* Wl  = (const float*)d_in[5];
    const float* g1  = (const float*)d_in[6];
    const float* be1 = (const float*)d_in[7];
    const float* W1  = (const float*)d_in[8];
    const float* b1  = (const float*)d_in[9];
    const float* W2  = (const float*)d_in[10];
    const float* b2  = (const float*)d_in[11];
    const float* g2  = (const float*)d_in[12];
    const float* be2 = (const float*)d_in[13];
    float* out = (float*)d_out;

    // Workspace layout (bytes)
    char* ws = (char*)d_ws;
    unsigned short* srcB = (unsigned short*)(ws + 0LL);          // 16 MB
    unsigned short* tgtB = (unsigned short*)(ws + 16777216LL);   // 16 MB -> attnB
    unsigned short* qT   = (unsigned short*)(ws + 33554432LL);   // 16 MB -> ln1B
    unsigned short* vB   = (unsigned short*)(ws + 50331648LL);   // 16 MB
    float*          kF   = (float*)(ws + 67108864LL);            // 33.5 MB -> msg
    unsigned short* ktT  = (unsigned short*)(ws + 100663296LL);  // 16 MB
    float*          ctxp = (float*)(ws + 117440512LL);           // 8 MB
    unsigned short* ctxT = (unsigned short*)(ws + 125829120LL);  // 512 KB
    unsigned short* WqT  = (unsigned short*)(ws + 126353408LL);
    unsigned short* WkT  = (unsigned short*)(ws + 126484480LL);
    unsigned short* WvT  = (unsigned short*)(ws + 126615552LL);
    unsigned short* WlT  = (unsigned short*)(ws + 126746624LL);
    unsigned short* W1T  = (unsigned short*)(ws + 126877696LL);  // 2 MB
    unsigned short* W2T  = (unsigned short*)(ws + 128974848LL);  // 1 MB
    float*          pm   = (float*)(ws + 130023424LL);
    float*          ps   = (float*)(ws + 130154496LL);
    float*          gm   = (float*)(ws + 130285568LL);
    float*          gi   = (float*)(ws + 130289664LL);
    unsigned short* hid  = (unsigned short*)(ws + 130293760LL);  // 16 MB

    unsigned short* attnB = tgtB;   // tgtB dead after K/V GEMMs
    unsigned short* ln1B  = qT;     // qT dead after context GEMM
    float*          msg   = kF;     // kF dead after softmax

    const long long LCe = (long long)LTOT * CDIM;   // 2097152 elements
    dim3 blk(256);

    // 0) fp32 -> bf16 conversions (activations flat, weights transposed)
    cvt_bf16<<<dim3(4096), blk, 0, stream>>>(src, srcB, (long long)MTOT * CDIM);
    cvt_bf16<<<dim3(4096), blk, 0, stream>>>(tgt, tgtB, (long long)MTOT * CDIM);
    cvt_bf16_t<<<dim3(8, 8),   blk, 0, stream>>>(Wq, WqT, CDIM, CDIM);
    cvt_bf16_t<<<dim3(8, 8),   blk, 0, stream>>>(Wk, WkT, CDIM, CDIM);
    cvt_bf16_t<<<dim3(8, 8),   blk, 0, stream>>>(Wv, WvT, CDIM, CDIM);
    cvt_bf16_t<<<dim3(8, 8),   blk, 0, stream>>>(Wl, WlT, CDIM, CDIM);
    cvt_bf16_t<<<dim3(16, 64), blk, 0, stream>>>(W1, W1T, 2 * CDIM, FFNH);
    cvt_bf16_t<<<dim3(64, 8),  blk, 0, stream>>>(W2, W2T, FFNH, CDIM);

    // 1) Q = src@Wq stored transposed per batch (bf16 [b][d][tok])
    gemm_bf16<0,0,0,2><<<dim3(4, 256, 1), blk, 0, stream>>>(
        srcB, nullptr, WqT, nullptr, qT, MTOT, CDIM, CDIM,
        CDIM, 0, CDIM, CDIM, 0, 0, 0, 1, 0, 0);
    // 2) K = tgt@Wk (f32, for softmax)
    gemm_bf16<0,0,0,0><<<dim3(4, 256, 1), blk, 0, stream>>>(
        tgtB, nullptr, WkT, nullptr, kF, MTOT, CDIM, CDIM,
        CDIM, 0, CDIM, CDIM, 0, 0, 0, 1, 0, 0);
    // 3) V = tgt@Wv (bf16)
    gemm_bf16<0,0,0,1><<<dim3(4, 256, 1), blk, 0, stream>>>(
        tgtB, nullptr, WvT, nullptr, vB, MTOT, CDIM, CDIM,
        CDIM, 0, CDIM, CDIM, 0, 0, 0, 1, 0, 0);

    // 4) softmax over tokens -> kt^T bf16 [b][c][tok]
    softmax_pass1<<<dim3(BBATCH, SM_CHUNKS), blk, 0, stream>>>(kF, pm, ps);
    softmax_pass2<<<dim3(BBATCH), blk, 0, stream>>>(pm, ps, gm, gi);
    softmax_pass3<<<dim3(BBATCH, SM_CHUNKS), blk, 0, stream>>>(kF, gm, gi, ktT);

    // 5) context[b] = q[b]^T @ ksm[b]  (A=qT [d][tok], B=ktT [e][tok], K=8192)
    gemm_bf16<0,0,1,0><<<dim3(4, 2, BBATCH * 8), blk, 0, stream>>>(
        qT, nullptr, ktT, nullptr, ctxp, CDIM, CDIM, LTOT,
        LTOT, 0, LTOT, CDIM, (long long)CDIM * LTOT, (long long)CDIM * LTOT,
        65536LL, 8, (long long)BBATCH * 65536LL, 0);
    reduce_split_bf16<<<dim3(1024), blk, 0, stream>>>(
        ctxp, ctxT, (long long)BBATCH * 65536LL, 8);

    // 6) attn[b] = v[b] @ ctx[b]^T  (B = ctxT row-major [d][e] == [N][K])
    gemm_bf16<0,0,0,1><<<dim3(4, 64, BBATCH), blk, 0, stream>>>(
        vB, nullptr, ctxT, nullptr, attnB, LTOT, CDIM, CDIM,
        CDIM, 0, CDIM, CDIM, LCe, 65536LL, LCe, 1, 0, 0);

    // 7) msg0 = attn @ Wl (f32)
    gemm_bf16<0,0,0,0><<<dim3(4, 256, 1), blk, 0, stream>>>(
        attnB, nullptr, WlT, nullptr, msg, MTOT, CDIM, CDIM,
        CDIM, 0, CDIM, CDIM, 0, 0, 0, 1, 0, 0);

    // 8) ln1 = LayerNorm(msg0) -> bf16
    layernorm_k<1,0><<<dim3(MTOT), blk, 0, stream>>>(msg, g1, be1, nullptr, ln1B);

    // 9) FFN chunked over M (4096 rows), hidden kept bf16
    for (int c8 = 0; c8 < 8; ++c8) {
        const long long mOff = (long long)c8 * 4096;
        gemm_bf16<1,2,0,1><<<dim3(32, 32, 1), blk, 0, stream>>>(
            srcB + mOff * CDIM, ln1B + mOff * CDIM, W1T, b1, hid,
            4096, FFNH, 2 * CDIM, CDIM, CDIM, 2 * CDIM, FFNH,
            0, 0, 0, 1, 0, CDIM);
        gemm_bf16<0,1,0,0><<<dim3(4, 32, 1), blk, 0, stream>>>(
            hid, nullptr, W2T, b2, msg + mOff * CDIM,
            4096, CDIM, FFNH, FFNH, 0, FFNH, CDIM,
            0, 0, 0, 1, 0, 0);
    }

    // 10) out = src + LayerNorm(msg1)
    layernorm_k<0,1><<<dim3(MTOT), blk, 0, stream>>>(msg, g2, be2, src, out);
}
